// DenseGATLayer_50397146251480
// MI455X (gfx1250) — matrix operand order repaired and checked by reading, not compiled, not run on hardware
//
#include <hip/hip_runtime.h>
#include <math.h>

// ---- sizes ----
#define BB 256
#define NN 192
#define DD 512
#define HH 8
#define HD 64
#define RR 16
#define ROWS (BB * NN)          // 49152
#define DFF  (4 * DD)           // 2048

typedef __attribute__((ext_vector_type(16))) _Float16 v16h;
typedef __attribute__((ext_vector_type(8)))  _Float16 v8h;
typedef __attribute__((ext_vector_type(8)))  float    v8f;
typedef __attribute__((ext_vector_type(4)))  float    v4f;

union hv16 { v16h v; v8h h[2]; };

__device__ __forceinline__ v8f wmma16x16x32(const hv16& a, const hv16& b, v8f c) {
  // v_wmma_f32_16x16x32_f16: (neg_a, A, neg_b, B, c_mod, C, reuse_a, reuse_b)
  return __builtin_amdgcn_wmma_f32_16x16x32_f16(false, a.v, false, b.v,
                                                (short)0, c, false, false);
}

// CDNA5 async global->LDS copy (16B per lane), tracked by ASYNCcnt.
// VDST = per-lane LDS byte address (low 32 bits of the generic pointer),
// VADDR = per-lane 64-bit global address, saddr off.
__device__ __forceinline__ void async_ld_b128(const _Float16* gsrc, _Float16* ldst) {
  unsigned l = (unsigned)(unsigned long long)(uintptr_t)ldst;
  unsigned long long g = (unsigned long long)(uintptr_t)gsrc;
  asm volatile("global_load_async_to_lds_b128 %0, %1, off"
               :: "v"(l), "v"(g) : "memory");
}
__device__ __forceinline__ void wait_async_le4() {
  asm volatile("s_wait_asynccnt 0x4" ::: "memory");
}
__device__ __forceinline__ void wait_async_le0() {
  asm volatile("s_wait_asynccnt 0x0" ::: "memory");
}

// ------------------------------------------------------------------
// f32 -> f16 transpose: out[n*K + k] = (f16) in[k*Nc + n]
// ------------------------------------------------------------------
__global__ __launch_bounds__(256)
void cvt_tr(const float* __restrict__ in, _Float16* __restrict__ out, int K, int Nc) {
  __shared__ float t[32][33];
  int nb = blockIdx.x * 32, kb = blockIdx.y * 32;
  int tx = threadIdx.x & 31, ty = threadIdx.x >> 5;   // 32 x 8
#pragma unroll
  for (int i = 0; i < 32; i += 8)
    t[ty + i][tx] = in[(size_t)(kb + ty + i) * Nc + nb + tx];
  __syncthreads();
#pragma unroll
  for (int i = 0; i < 32; i += 8)
    out[(size_t)(nb + ty + i) * K + kb + tx] = (_Float16)t[tx][ty + i];
}

// ------------------------------------------------------------------
// LayerNorm: one wave per row of 512, f32 in -> f16 out
// ------------------------------------------------------------------
__global__ __launch_bounds__(256)
void ln_kernel(const float* __restrict__ x, const float* __restrict__ gam,
               const float* __restrict__ bet, _Float16* __restrict__ out, int rows) {
  int row = blockIdx.x * 8 + (threadIdx.x >> 5);
  if (row >= rows) return;
  int lane = threadIdx.x & 31;
  const float* xr = x + (size_t)row * DD;
  v4f vals[4];
  float s = 0.f, ss = 0.f;
#pragma unroll
  for (int i = 0; i < 4; ++i) {
    vals[i] = *(const v4f*)(xr + lane * 16 + i * 4);
#pragma unroll
    for (int j = 0; j < 4; ++j) { s += vals[i][j]; ss += vals[i][j] * vals[i][j]; }
  }
#pragma unroll
  for (int d = 1; d < 32; d <<= 1) {
    s  += __shfl_xor(s,  d, 32);
    ss += __shfl_xor(ss, d, 32);
  }
  float mu   = s  * (1.0f / (float)DD);
  float var  = ss * (1.0f / (float)DD) - mu * mu;
  float rstd = rsqrtf(var + 1e-5f);
  _Float16* orow = out + (size_t)row * DD;
#pragma unroll
  for (int i = 0; i < 4; ++i)
#pragma unroll
    for (int j = 0; j < 4; ++j) {
      int c = lane * 16 + i * 4 + j;
      orow[c] = (_Float16)((vals[i][j] - mu) * rstd * gam[c] + bet[c]);
    }
}

// ------------------------------------------------------------------
// WMMA GEMM: C[M,Nc] = A[M,K](f16) * Wt^T + bias, where Wt is the
// PRE-TRANSPOSED weight [Nc,K] (f16). 64x64 tile, 4 waves / block.
// Tiles staged via CDNA5 async global->LDS b128, double-buffered:
// prefetch tile j+1 while WMMAs consume tile j (ASYNCcnt discipline).
// EPI: 0 = store f16, 1 = exact GELU then store f16,
//      2 = +residual(f32) then store f32
// ------------------------------------------------------------------
template <int EPI>
__global__ __launch_bounds__(128)
void gemm_wmma(const _Float16* __restrict__ A, const _Float16* __restrict__ Wt,
               const float* __restrict__ bias, int M, int K, int Nc,
               _Float16* __restrict__ outH, const float* __restrict__ resF,
               float* __restrict__ outF) {
  __shared__ _Float16 As[2][64][40];   // [buf][m][k], padded rows (80B)
  __shared__ _Float16 Bs[2][64][40];   // [buf][n][k], padded rows

  const int tid  = threadIdx.x;
  const int lane = tid & 31;
  const int wave = tid >> 5;
  const int hi   = lane >> 4;   // 0 for lanes 0-15, 1 for 16-31
  const int ln   = lane & 15;
  const int m0 = blockIdx.y * 64, n0 = blockIdx.x * 64;

  // staging role: thread t owns 16 contiguous halves of one 64x32 tile row
  const int rS  = tid >> 1;
  const int ksS = (tid & 1) * 16;

  auto stage = [&](int buf, int k0) {
    const _Float16* ga = A  + (size_t)(m0 + rS) * K + k0 + ksS;
    const _Float16* gb = Wt + (size_t)(n0 + rS) * K + k0 + ksS;
    async_ld_b128(ga,     &As[buf][rS][ksS]);
    async_ld_b128(ga + 8, &As[buf][rS][ksS + 8]);
    async_ld_b128(gb,     &Bs[buf][rS][ksS]);
    async_ld_b128(gb + 8, &Bs[buf][rS][ksS + 8]);
  };

  v8f acc[4] = {};
  const int nk = K >> 5;          // K/32 k-steps
  stage(0, 0);

  for (int j = 0; j < nk; ++j) {
    const int buf = j & 1;
    if (j + 1 < nk) {
      stage(buf ^ 1, (j + 1) << 5);
      wait_async_le4();           // tile j landed; tile j+1 still in flight
    } else {
      wait_async_le0();
    }
    __syncthreads();

    // A fragment (16x32): lane row = wave*16+ln, chunks at hi*8 and hi*8+16
    hv16 af;
    {
      int r = wave * 16 + ln;
      af.h[0] = *(const v8h*)&As[buf][r][hi * 8];
      af.h[1] = *(const v8h*)&As[buf][r][hi * 8 + 16];
    }
#pragma unroll
    for (int nt = 0; nt < 4; ++nt) {
      hv16 bf;
      int c = nt * 16 + ln;
      bf.h[0] = *(const v8h*)&Bs[buf][c][hi * 8];
      bf.h[1] = *(const v8h*)&Bs[buf][c][hi * 8 + 16];
      acc[nt] = wmma16x16x32(af, bf, acc[nt]);
    }
    __syncthreads();              // protect buf from next prefetch overwrite
  }

  // epilogue: C element (m = r + 8*hi, n = ln) per 16x16 tile
#pragma unroll
  for (int nt = 0; nt < 4; ++nt) {
    int n = n0 + nt * 16 + ln;
    float bv = bias[n];
#pragma unroll
    for (int r = 0; r < 8; ++r) {
      int m = m0 + wave * 16 + r + 8 * hi;
      float vsum = acc[nt][r] + bv;
      size_t idx = (size_t)m * Nc + n;
      if (EPI == 0) {
        outH[idx] = (_Float16)vsum;
      } else if (EPI == 1) {
        float g = 0.5f * vsum * (1.0f + erff(vsum * 0.70710678118654752f));
        outH[idx] = (_Float16)g;
      } else {
        outF[idx] = vsum + resF[idx];
      }
    }
  }
}

// ------------------------------------------------------------------
// Attention: one block (128 thr = 4 waves) per (b,h).
// Wave handles 3 of the 12 row-tiles: QK^T (24 wmma) -> masked softmax
// (half-wave shfl reductions) -> P staged via LDS -> PV (24 wmma).
// ------------------------------------------------------------------
__global__ __launch_bounds__(128)
void attn_kernel(const _Float16* __restrict__ q, const _Float16* __restrict__ k,
                 const _Float16* __restrict__ v, const float* __restrict__ adj,
                 const int* __restrict__ etypes, const float* __restrict__ relb,
                 _Float16* __restrict__ attnOut) {
  const int bh = blockIdx.x;
  const int b  = bh >> 3;          // / HH
  const int h  = bh & 7;           // % HH
  const int tid  = threadIdx.x;
  const int lane = tid & 31;
  const int wave = tid >> 5;
  const int hi   = lane >> 4;
  const int ln   = lane & 15;

  __shared__ _Float16 Vt[64][200];        // V^T  [col][k]
  __shared__ _Float16 Pb[4][16][200];     // per-wave P tile [m][n]

  // stage V^T: V[b, kr, h*64 + c] -> Vt[c][kr]
  for (int idx = tid; idx < NN * 8; idx += 128) {
    int kr = idx >> 3;
    int c0 = (idx & 7) * 8;
    v8h vv = *(const v8h*)(v + (size_t)(b * NN + kr) * DD + h * HD + c0);
#pragma unroll
    for (int i = 0; i < 8; ++i) Vt[c0 + i][kr] = vv[i];
  }
  __syncthreads();

  const float scale = 0.125f;  // HD^-0.5

  for (int mi = 0; mi < 3; ++mi) {
    int mt   = wave + mi * 4;    // 0..11
    int mrow = mt * 16;

    // Q A-fragments for both 32-wide k-steps
    hv16 qf[2];
    {
      size_t base = (size_t)(b * NN + mrow + ln) * DD + h * HD;
#pragma unroll
      for (int s = 0; s < 2; ++s) {
        qf[s].h[0] = *(const v8h*)(q + base + s * 32 + hi * 8);
        qf[s].h[1] = *(const v8h*)(q + base + s * 32 + hi * 8 + 16);
      }
    }

    // scores: 12 n-tiles x (k=64 in 2 steps)
    v8f acc[12] = {};
#pragma unroll
    for (int nt = 0; nt < 12; ++nt) {
      size_t kb = (size_t)(b * NN + nt * 16 + ln) * DD + h * HD + hi * 8;
#pragma unroll
      for (int s = 0; s < 2; ++s) {
        hv16 kf;
        kf.h[0] = *(const v8h*)(k + kb + s * 32);
        kf.h[1] = *(const v8h*)(k + kb + s * 32 + 16);
        acc[nt] = wmma16x16x32(qf[s], kf, acc[nt]);
      }
    }

    // scale + rel_bias + adjacency mask; track row max
    float rmax[8];
#pragma unroll
    for (int r = 0; r < 8; ++r) rmax[r] = -INFINITY;
#pragma unroll
    for (int nt = 0; nt < 12; ++nt) {
      int n = nt * 16 + ln;
#pragma unroll
      for (int r = 0; r < 8; ++r) {
        int m = mrow + r + 8 * hi;
        size_t ei = (size_t)(b * NN + m) * NN + n;
        float a = adj[ei];
        float sc = acc[nt][r] * scale + relb[etypes[ei] * HH + h];
        sc = (a != 0.0f) ? sc : -INFINITY;
        acc[nt][r] = sc;
        rmax[r] = fmaxf(rmax[r], sc);
      }
    }
    // reduce across the 16 lanes of each half-wave (rows stay in-group)
#pragma unroll
    for (int r = 0; r < 8; ++r)
#pragma unroll
      for (int d = 1; d < 16; d <<= 1)
        rmax[r] = fmaxf(rmax[r], __shfl_xor(rmax[r], d, 32));

    float rsum[8];
#pragma unroll
    for (int r = 0; r < 8; ++r) rsum[r] = 0.0f;
#pragma unroll
    for (int nt = 0; nt < 12; ++nt)
#pragma unroll
      for (int r = 0; r < 8; ++r) {
        float p = (acc[nt][r] == -INFINITY) ? 0.0f : __expf(acc[nt][r] - rmax[r]);
        acc[nt][r] = p;
        rsum[r] += p;
      }
#pragma unroll
    for (int r = 0; r < 8; ++r)
#pragma unroll
      for (int d = 1; d < 16; d <<= 1)
        rsum[r] += __shfl_xor(rsum[r], d, 32);

    float rinv[8];
#pragma unroll
    for (int r = 0; r < 8; ++r) rinv[r] = (rsum[r] > 0.0f) ? (1.0f / rsum[r]) : 0.0f; // nan_to_num

    // P -> LDS (C layout -> row-major tile)
#pragma unroll
    for (int nt = 0; nt < 12; ++nt)
#pragma unroll
      for (int r = 0; r < 8; ++r) {
        int m = r + 8 * hi;
        Pb[wave][m][nt * 16 + ln] = (_Float16)(acc[nt][r] * rinv[r]);
      }
    __syncthreads();

    // O = P(16x192) * V(192x64): 6 k-steps x 4 n-tiles
    v8f oacc[4] = {};
#pragma unroll
    for (int s2 = 0; s2 < 6; ++s2) {
      hv16 pf;
      pf.h[0] = *(const v8h*)&Pb[wave][ln][s2 * 32 + hi * 8];
      pf.h[1] = *(const v8h*)&Pb[wave][ln][s2 * 32 + hi * 8 + 16];
#pragma unroll
      for (int nt2 = 0; nt2 < 4; ++nt2) {
        hv16 vf;
        vf.h[0] = *(const v8h*)&Vt[nt2 * 16 + ln][s2 * 32 + hi * 8];
        vf.h[1] = *(const v8h*)&Vt[nt2 * 16 + ln][s2 * 32 + hi * 8 + 16];
        oacc[nt2] = wmma16x16x32(pf, vf, oacc[nt2]);
      }
    }
    // store O tile to [B,N,D] f16 (head-sliced columns)
#pragma unroll
    for (int nt2 = 0; nt2 < 4; ++nt2)
#pragma unroll
      for (int r = 0; r < 8; ++r) {
        int m = mrow + r + 8 * hi;
        attnOut[(size_t)(b * NN + m) * DD + h * HD + nt2 * 16 + ln] =
            (_Float16)oacc[nt2][r];
      }
    __syncthreads();
  }
}

// ------------------------------------------------------------------
// launch
// ------------------------------------------------------------------
extern "C" void kernel_launch(void* const* d_in, const int* in_sizes, int n_in,
                              void* d_out, int out_size, void* d_ws, size_t ws_size,
                              hipStream_t stream) {
  (void)in_sizes; (void)n_in; (void)out_size; (void)ws_size;
  const float* x     = (const float*)d_in[0];
  const float* adj   = (const float*)d_in[1];
  const int*   et    = (const int*)d_in[2];
  const float* wq    = (const float*)d_in[3];
  const float* bq    = (const float*)d_in[4];
  const float* wk    = (const float*)d_in[5];
  const float* bk    = (const float*)d_in[6];
  const float* wv    = (const float*)d_in[7];
  const float* bv    = (const float*)d_in[8];
  const float* wo    = (const float*)d_in[9];
  const float* bo    = (const float*)d_in[10];
  const float* relb  = (const float*)d_in[11];
  const float* ln1g  = (const float*)d_in[12];
  const float* ln1b  = (const float*)d_in[13];
  const float* ln2g  = (const float*)d_in[14];
  const float* ln2b  = (const float*)d_in[15];
  const float* w1    = (const float*)d_in[16];
  const float* b1    = (const float*)d_in[17];
  const float* w2    = (const float*)d_in[18];
  const float* b2    = (const float*)d_in[19];

  char* ws = (char*)d_ws;
  const size_t SZ16 = (size_t)ROWS * DD * sizeof(_Float16);   // 50,331,648
  _Float16* hH  = (_Float16*)(ws);                 // LN1 out; reused as LN2 out
  _Float16* qH  = (_Float16*)(ws + 1 * SZ16);
  _Float16* kH  = (_Float16*)(ws + 2 * SZ16);
  _Float16* vH  = (_Float16*)(ws + 3 * SZ16);
  _Float16* aH  = (_Float16*)(ws + 4 * SZ16);
  _Float16* f1H = (_Float16*)(ws + 1 * SZ16);      // reuses q/k/v/attn region (4*SZ16)
  float*    oF  = (float*)(ws + 5 * SZ16);         // f32 residual stream (2*SZ16 bytes)
  char* wbase   = ws + 7 * SZ16;
  _Float16* wqT = (_Float16*)(wbase);                        // [512,512]   transposed
  _Float16* wkT = (_Float16*)(wbase + 1 * 524288);
  _Float16* wvT = (_Float16*)(wbase + 2 * 524288);
  _Float16* woT = (_Float16*)(wbase + 3 * 524288);
  _Float16* w1T = (_Float16*)(wbase + 4 * 524288);           // [2048,512]
  _Float16* w2T = (_Float16*)(wbase + 4 * 524288 + 2097152); // [512,2048]

  // 1) weights -> f16, transposed to [Nc,K] so GEMM B-tiles are k-contiguous
  dim3 gT512(16, 16);
  cvt_tr<<<gT512, 256, 0, stream>>>(wq, wqT, DD, DD);
  cvt_tr<<<gT512, 256, 0, stream>>>(wk, wkT, DD, DD);
  cvt_tr<<<gT512, 256, 0, stream>>>(wv, wvT, DD, DD);
  cvt_tr<<<gT512, 256, 0, stream>>>(wo, woT, DD, DD);
  cvt_tr<<<dim3(DFF / 32, DD / 32), 256, 0, stream>>>(w1, w1T, DD, DFF);   // -> [2048,512]
  cvt_tr<<<dim3(DD / 32, DFF / 32), 256, 0, stream>>>(w2, w2T, DFF, DD);   // -> [512,2048]

  // 2) LN1
  ln_kernel<<<ROWS / 8, 256, 0, stream>>>(x, ln1g, ln1b, hH, ROWS);

  // 3) Q,K,V projections
  dim3 gP(DD / 64, ROWS / 64);   // (8, 768)
  gemm_wmma<0><<<gP, 128, 0, stream>>>(hH, wqT, bq, ROWS, DD, DD, qH, nullptr, nullptr);
  gemm_wmma<0><<<gP, 128, 0, stream>>>(hH, wkT, bk, ROWS, DD, DD, kH, nullptr, nullptr);
  gemm_wmma<0><<<gP, 128, 0, stream>>>(hH, wvT, bv, ROWS, DD, DD, vH, nullptr, nullptr);

  // 4) attention
  attn_kernel<<<BB * HH, 128, 0, stream>>>(qH, kH, vH, adj, et, relb, aH);

  // 5) output projection + residual -> oF (f32)
  gemm_wmma<2><<<gP, 128, 0, stream>>>(aH, woT, bo, ROWS, DD, DD, nullptr, x, oF);

  // 6) LN2 -> hH
  ln_kernel<<<ROWS / 8, 256, 0, stream>>>(oF, ln2g, ln2b, hH, ROWS);

  // 7) FFN1 + exact GELU
  dim3 gF1(DFF / 64, ROWS / 64); // (32, 768)
  gemm_wmma<1><<<gF1, 128, 0, stream>>>(hH, w1T, b1, ROWS, DD, DFF, f1H, nullptr, nullptr);

  // 8) FFN2 + residual -> d_out (f32)
  gemm_wmma<2><<<gP, 128, 0, stream>>>(f1H, w2T, b2, ROWS, DFF, DD, nullptr, oF, (float*)d_out);
}
